// LSTMModel_2310692405872
// MI455X (gfx1250) — compile-verified, matrix operand order checked by reading, not run on hardware
//
#include <hip/hip_runtime.h>

// CDNA5 / gfx1250 fused 3-layer LSTM + FC using f16 WMMA with f32 accumulation.
//
// Strategy: batch is the only parallel dimension (T is a recurrence). Each
// 128-thread workgroup (4 wave32) owns a 16-row batch tile and runs all 3
// layers fused per timestep:
//   z[16,256] = [x|h][16,128] @ W^T[128,256]   (4 gates x 64-padded N)
// Each wave computes one 16-wide N subtile for ALL four gates (16 WMMAs of
// 16x16x32 f16), so sigmoid/tanh gating and the c-state update stay entirely
// in that wave's registers. h is exchanged between layers via LDS in f16.
// Weights are converted to f16 and zero-padded into LDS once; zero padding
// guarantees pad lanes never affect real outputs.

typedef __attribute__((ext_vector_type(16))) _Float16 v16h;
typedef __attribute__((ext_vector_type(8)))  float    v8f;

namespace lstmcfg {
constexpr int H_      = 50;
constexpr int T_      = 512;
constexpr int B_      = 2048;
constexpr int BT      = 16;    // batch rows per workgroup (one WMMA M tile)
constexpr int KPAD    = 128;   // 64 (input pad) + 64 (hidden pad)
constexpr int NPAD    = 256;   // 4 gates * 64
constexpr int NLAY    = 3;
constexpr int THREADS = 128;   // 4 wave32

constexpr int W_ELEMS    = NLAY * NPAD * KPAD;   // f16 weights
constexpr int W_BYTES    = W_ELEMS * 2;          // 196608
constexpr int BIAS_ELEMS = NLAY * NPAD;          // f32 bias
constexpr int BIAS_BYTES = BIAS_ELEMS * 4;       // 3072
constexpr int XH_ELEMS   = NLAY * BT * KPAD;     // f16 [x|h] per layer
constexpr int XH_BYTES   = XH_ELEMS * 2;         // 12288
constexpr int XB_ELEMS   = BT * T_;              // f16 staged x slice
constexpr int XB_BYTES   = XB_ELEMS * 2;         // 16384
constexpr size_t LDS_BYTES =
    (size_t)W_BYTES + BIAS_BYTES + XH_BYTES + XB_BYTES;  // 228352 < 320K
}  // namespace lstmcfg

__device__ __forceinline__ float sigmoid_fast(float x) {
  return 1.0f / (1.0f + __expf(-x));
}
__device__ __forceinline__ float tanh_fast(float x) {
  float e = __expf(2.0f * x);
  return (e - 1.0f) / (e + 1.0f);
}

__global__ __launch_bounds__(lstmcfg::THREADS)
void lstm3_wmma_kernel(const float* __restrict__ x,
                       const float* __restrict__ wih1, const float* __restrict__ whh1,
                       const float* __restrict__ b1,
                       const float* __restrict__ wih2, const float* __restrict__ whh2,
                       const float* __restrict__ b2,
                       const float* __restrict__ wih3, const float* __restrict__ whh3,
                       const float* __restrict__ b3,
                       const float* __restrict__ wfc, const float* __restrict__ bfc,
                       float* __restrict__ out)
{
  using namespace lstmcfg;
  extern __shared__ v16h smem_v[];              // 32-byte aligned base
  char* smem = reinterpret_cast<char*>(smem_v);
  _Float16* Wlds = reinterpret_cast<_Float16*>(smem);                       // [3][256][128]
  float*    Blds = reinterpret_cast<float*>(smem + W_BYTES);                // [3][256]
  _Float16* XH   = reinterpret_cast<_Float16*>(smem + W_BYTES + BIAS_BYTES);// [3][16][128]
  _Float16* XB   = reinterpret_cast<_Float16*>(smem + W_BYTES + BIAS_BYTES + XH_BYTES); // [16][512]

  const int tid  = threadIdx.x;
  const int wave = tid >> 5;
  const int lane = tid & 31;
  const int b0   = blockIdx.x * BT;

  const float* wih[NLAY] = { wih1, wih2, wih3 };
  const float* whh[NLAY] = { whh1, whh2, whh3 };
  const float* bb [NLAY] = { b1, b2, b3 };

  // ---- stage weights once: fp32 -> f16, zero-padded, [layer][N][K], N-major ----
  for (int idx = tid; idx < W_ELEMS; idx += THREADS) {
    const int l   = idx / (NPAD * KPAD);
    const int rem = idx - l * (NPAD * KPAD);
    const int n   = rem >> 7;       // KPAD = 128
    const int k   = rem & 127;
    const int g   = n >> 6;
    const int j   = n & 63;
    float v = 0.0f;
    if (j < H_) {
      const int r = g * H_ + j;     // original gate-major row in [4H, .]
      if (k < 64) {                 // input-weight half
        const int inw = (l == 0) ? 1 : H_;
        if (k < inw) v = wih[l][r * inw + k];
      } else {                      // hidden-weight half
        const int kh = k - 64;
        if (kh < H_) v = whh[l][r * H_ + kh];
      }
    }
    Wlds[idx] = (_Float16)v;
  }
  for (int idx = tid; idx < BIAS_ELEMS; idx += THREADS) {
    const int l = idx / NPAD;
    const int n = idx - l * NPAD;
    const int g = n >> 6, j = n & 63;
    Blds[idx] = (j < H_) ? bb[l][g * H_ + j] : 0.0f;
  }
  // stage this block's x slice [BT][T] (coalesced along T)
  for (int idx = tid; idx < XB_ELEMS; idx += THREADS) {
    const int r = idx >> 9;         // T = 512
    const int t = idx & 511;
    XB[idx] = (_Float16)x[(size_t)(b0 + r) * T_ + t];
  }
  // zero [x|h] buffers: h0 = 0, input pads = 0
  for (int idx = tid; idx < XH_ELEMS; idx += THREADS) XH[idx] = (_Float16)0.0f;
  __syncthreads();

  // wave -> one 16-wide N subtile (of the 64-padded gate dim), all 4 gates
  const int nbase = wave * 16;
  const int rowA  = lane & 15;      // A: M = lane (lo/hi half pick K chunk)
  const int hiK   = lane >> 4;      // A/B: second 16-K half; C/D: M += 8
  const int colh  = nbase + (lane & 15);  // this lane's N column

  float bias_v[NLAY][4];
  #pragma unroll
  for (int l = 0; l < NLAY; ++l)
    #pragma unroll
    for (int g = 0; g < 4; ++g)
      bias_v[l][g] = Blds[l * NPAD + g * 64 + colh];

  v8f cst[NLAY];
  #pragma unroll
  for (int l = 0; l < NLAY; ++l) { v8f z = {}; cst[l] = z; }

  for (int t = 0; t < T_; ++t) {
    if (tid < BT) XH[tid * KPAD] = XB[tid * T_ + t];   // layer-0 input, col 0
    __syncthreads();

    #pragma unroll
    for (int l = 0; l < NLAY; ++l) {
      const _Float16* xh = XH + l * BT * KPAD;
      // A fragments: 4 K-chunks of 32 (contiguous 32B per lane)
      v16h a0 = *(const v16h*)(xh + rowA * KPAD +  0 + hiK * 16);
      v16h a1 = *(const v16h*)(xh + rowA * KPAD + 32 + hiK * 16);
      v16h a2 = *(const v16h*)(xh + rowA * KPAD + 64 + hiK * 16);
      v16h a3 = *(const v16h*)(xh + rowA * KPAD + 96 + hiK * 16);

      const _Float16* Wl = Wlds + l * NPAD * KPAD;
      v8f acc[4];
      #pragma unroll
      for (int g = 0; g < 4; ++g) {
        const _Float16* wr = Wl + (g * 64 + colh) * KPAD;  // weight row = this lane's N col
        v16h bf0 = *(const v16h*)(wr +  0 + hiK * 16);
        v16h bf1 = *(const v16h*)(wr + 32 + hiK * 16);
        v16h bf2 = *(const v16h*)(wr + 64 + hiK * 16);
        v16h bf3 = *(const v16h*)(wr + 96 + hiK * 16);
        v8f a_ = {};
        a_ = __builtin_amdgcn_wmma_f32_16x16x32_f16(false, a0, false, bf0, (short)0, a_, false, false);
        a_ = __builtin_amdgcn_wmma_f32_16x16x32_f16(false, a1, false, bf1, (short)0, a_, false, false);
        a_ = __builtin_amdgcn_wmma_f32_16x16x32_f16(false, a2, false, bf2, (short)0, a_, false, false);
        a_ = __builtin_amdgcn_wmma_f32_16x16x32_f16(false, a3, false, bf3, (short)0, a_, false, false);
        acc[g] = a_;
      }

      // gating entirely in-register: i,f,g,o for the same (row, col) live here
      v8f c = cst[l];
      _Float16 hv[8];
      #pragma unroll
      for (int v = 0; v < 8; ++v) {
        const float zi = acc[0][v] + bias_v[l][0];
        const float zf = acc[1][v] + bias_v[l][1];
        const float zg = acc[2][v] + bias_v[l][2];
        const float zo = acc[3][v] + bias_v[l][3];
        const float ig = sigmoid_fast(zi);
        const float fg = sigmoid_fast(zf);
        const float gg = tanh_fast(zg);
        const float og = sigmoid_fast(zo);
        const float cv = fg * c[v] + ig * gg;
        c[v] = cv;
        hv[v] = (_Float16)(og * tanh_fast(cv));
      }
      cst[l] = c;

      __syncthreads();  // all waves done READING this layer's [x|h] (and next layer's input)
      _Float16* xh_self = XH + l * BT * KPAD;
      _Float16* xh_next = (l < NLAY - 1) ? (XH + (l + 1) * BT * KPAD) : nullptr;
      #pragma unroll
      for (int v = 0; v < 8; ++v) {
        const int row = hiK * 8 + v;            // C/D row layout
        xh_self[row * KPAD + 64 + colh] = hv[v];          // own h(t) for t+1
        if (xh_next) xh_next[row * KPAD + colh] = hv[v];  // next layer's input at t
      }
      __syncthreads();
    }
  }

  // final FC: out[b] = h3_last . w_fc + b_fc
  if (tid < BT) {
    const _Float16* h3 = XH + 2 * BT * KPAD + tid * KPAD + 64;
    float s = bfc[0];
    #pragma unroll 10
    for (int j = 0; j < H_; ++j) s += (float)h3[j] * wfc[j];
    out[b0 + tid] = s;
  }
}

extern "C" void kernel_launch(void* const* d_in, const int* in_sizes, int n_in,
                              void* d_out, int out_size, void* d_ws, size_t ws_size,
                              hipStream_t stream) {
  using namespace lstmcfg;
  (void)in_sizes; (void)n_in; (void)out_size; (void)d_ws; (void)ws_size;

  const float* x    = (const float*)d_in[0];
  const float* wih1 = (const float*)d_in[1];
  const float* whh1 = (const float*)d_in[2];
  const float* b1   = (const float*)d_in[3];
  const float* wih2 = (const float*)d_in[4];
  const float* whh2 = (const float*)d_in[5];
  const float* b2   = (const float*)d_in[6];
  const float* wih3 = (const float*)d_in[7];
  const float* whh3 = (const float*)d_in[8];
  const float* b3   = (const float*)d_in[9];
  const float* wfc  = (const float*)d_in[10];
  const float* bfc  = (const float*)d_in[11];
  float* out = (float*)d_out;

  // LDS > 64 KB default: raise the dynamic-shared cap (not a stream op; capture-safe)
  hipFuncSetAttribute(reinterpret_cast<const void*>(lstm3_wmma_kernel),
                      hipFuncAttributeMaxDynamicSharedMemorySize, (int)LDS_BYTES);

  dim3 grid(B_ / BT);   // 128 workgroups -> one per WGP
  dim3 block(THREADS);  // 4 wave32
  lstm3_wmma_kernel<<<grid, block, LDS_BYTES, stream>>>(
      x, wih1, whh1, b1, wih2, whh2, b2, wih3, whh3, b3, wfc, bfc, out);
}
